// SelfMultiHeadAttention_23261542875442
// MI455X (gfx1250) — matrix operand order repaired and checked by reading, not compiled, not run on hardware
//
#include <hip/hip_runtime.h>

typedef __bf16 bf16;
typedef __attribute__((ext_vector_type(16))) __bf16 v16bf;
typedef __attribute__((ext_vector_type(8)))  __bf16 bf16x8;
typedef __attribute__((ext_vector_type(8)))  float  v8f;
typedef __attribute__((ext_vector_type(4)))  int    v4i;

#define NEG_BIG (-3.0e38f)

#if defined(__has_builtin)
#  if __has_builtin(__builtin_amdgcn_global_load_async_to_lds_b128) && \
      __has_builtin(__builtin_amdgcn_s_wait_asynccnt)
#    define USE_ASYNC_LDS 1
#  endif
#endif
#ifndef USE_ASYNC_LDS
#  define USE_ASYNC_LDS 0
#endif

__device__ __forceinline__ v16bf concat8(bf16x8 lo, bf16x8 hi) {
  return __builtin_shufflevector(lo, hi, 0, 1, 2, 3, 4, 5, 6, 7,
                                 8, 9, 10, 11, 12, 13, 14, 15);
}
__device__ __forceinline__ __attribute__((address_space(1))) v4i* gcast(const void* p) {
  return (__attribute__((address_space(1))) v4i*)(void*)(p);
}
__device__ __forceinline__ __attribute__((address_space(3))) v4i* lcast(void* p) {
  return (__attribute__((address_space(3))) v4i*)(p);
}
// one 16B global->LDS copy (async DMA when available, else through VGPRs)
__device__ __forceinline__ void cp16(void* lds, const void* glob) {
#if USE_ASYNC_LDS
  __builtin_amdgcn_global_load_async_to_lds_b128(gcast(glob), lcast(lds), 0, 0);
#else
  *(bf16x8*)lds = *(const bf16x8*)glob;
#endif
}
__device__ __forceinline__ void cp_wait() {
#if USE_ASYNC_LDS
  __builtin_amdgcn_s_wait_asynccnt(0);
#endif
}

// -------------------------------------------------- convert + transpose weights
// src[K][N] f32 (row-major) -> dst[N][K] bf16 (column-major) : one-time, L2 resident
__global__ __launch_bounds__(256) void cvt_transpose(const float* __restrict__ src,
                                                     bf16* __restrict__ dst,
                                                     int K, int N) {
  long i = (long)blockIdx.x * 256 + threadIdx.x;
  if (i < (long)K * N) {
    int k = (int)(i / N), n = (int)(i % N);       // coalesced read
    dst[(long)n * K + k] = (bf16)src[i];
  }
}

// ---------------------------------------------------------------- QKV GEMM
// C[8192,3072] = x @ Wqkv + b ; Q,K -> [bh][S][64] ; V -> [bh][64][S] (transposed)
#define TM 32
#define TN 128
#define KC 64
__global__ __launch_bounds__(256) void qkv_gemm(const float* __restrict__ x,
                                                const bf16* __restrict__ WqT, // [3072][1024]
                                                const float* __restrict__ bq,
                                                bf16* __restrict__ Qh,
                                                bf16* __restrict__ Kh,
                                                bf16* __restrict__ VhT) {
  __shared__ __align__(16) bf16 As[2][TM][KC + 8];   // 2 x 32 x 72
  __shared__ __align__(16) bf16 BsT[2][TN][KC + 8];  // 2 x 128 x 72 (N-major)
  const int tid  = threadIdx.x;
  const int lane = tid & 31, w = tid >> 5;
  const int wr = w >> 2, wc = w & 3;                 // 2 x 4 wave grid
  const int hi = lane >> 4, ln = lane & 15;
  const int kb = hi << 3, kb2 = hi << 4;
  const int m0 = blockIdx.x * TM;
  const int n0 = blockIdx.y * TN;

  auto stageA = [&](int k0, int bi) {                // 32x64 f32 -> bf16, 1 chunk/thr
    int r = tid >> 3, c = (tid & 7) * 8;
    const float4* xp = (const float4*)&x[(long)(m0 + r) * 1024 + k0 + c];
    float4 f0 = xp[0], f1 = xp[1];
    bf16x8 v;
    v[0] = (bf16)f0.x; v[1] = (bf16)f0.y; v[2] = (bf16)f0.z; v[3] = (bf16)f0.w;
    v[4] = (bf16)f1.x; v[5] = (bf16)f1.y; v[6] = (bf16)f1.z; v[7] = (bf16)f1.w;
    *(bf16x8*)&As[bi][r][c] = v;
  };
  auto issueB = [&](int k0, int bi) {                // 128 x 64 bf16 via async DMA
    #pragma unroll
    for (int it = 0; it < 4; ++it) {
      int i = tid + it * 256;
      int n = i >> 3, c = (i & 7) * 8;
      cp16(&BsT[bi][n][c], &WqT[(long)(n0 + n) * 1024 + k0 + c]);
    }
  };

  v8f acc[2] = {};
  stageA(0, 0);
  issueB(0, 0);
  int buf = 0;
  for (int k0 = 0; k0 < 1024; k0 += KC, buf ^= 1) {
    cp_wait();
    __syncthreads();
    #pragma unroll
    for (int kc = 0; kc < KC; kc += 32) {
      const bf16* ar = &As[buf][16 * wr + ln][kc + kb];
      v16bf a = concat8(*(const bf16x8*)ar, *(const bf16x8*)(ar + 16));
      #pragma unroll
      for (int nf = 0; nf < 2; ++nf) {
        const bf16* br = &BsT[buf][32 * wc + 16 * nf + ln][kc + kb];
        v16bf b = concat8(*(const bf16x8*)br, *(const bf16x8*)(br + 16));
        acc[nf] = __builtin_amdgcn_wmma_f32_16x16x32_bf16(false, a, false, b,
                                                          (short)0, acc[nf], false, false);
      }
    }
    __syncthreads();
    if (k0 + KC < 1024) {                            // stage next tile after compute; waited at loop top
      __builtin_prefetch(&x[(long)m0 * 1024 + k0 + 2 * KC], 0, 3);
      stageA(k0 + KC, buf ^ 1);
      issueB(k0 + KC, buf ^ 1);
    }
  }
  const int s0 = m0 + 16 * wr + (hi << 3);           // 8 consecutive tokens per lane
  const int bb = s0 >> 10, srow = s0 & 1023;
  #pragma unroll
  for (int nf = 0; nf < 2; ++nf) {
    const int col = n0 + 32 * wc + 16 * nf + ln;     // 0..3071
    const int which = col >> 10;
    const int d = col & 1023, h = d >> 6, dh = d & 63;
    const int bh = bb * 16 + h;
    const float bias = bq[col];
    if (which == 2) {                                // V transposed: contiguous tokens
      bf16x8 pk;
      #pragma unroll
      for (int r = 0; r < 8; ++r) pk[r] = (bf16)(acc[nf][r] + bias);
      *(bf16x8*)&VhT[(long)bh * 65536 + (long)dh * 1024 + srow] = pk;
    } else {
      bf16* dst = (which == 0) ? Qh : Kh;
      #pragma unroll
      for (int r = 0; r < 8; ++r)
        dst[((long)bh * 1024 + srow + r) * 64 + dh] = (bf16)(acc[nf][r] + bias);
    }
  }
}

// ---------------------------------------------------------------- attention
// one block per (b*h, 64 q rows); 4 waves x 16 q rows; flash online softmax, causal;
// double-buffered async K/V tiles: DMA(c+1) issued after compute(c), waited before compute(c+1)
__global__ __launch_bounds__(128) void attn_fwd(const bf16* __restrict__ Qh,
                                                const bf16* __restrict__ Kh,
                                                const bf16* __restrict__ VhT,
                                                bf16* __restrict__ Obuf) {
  __shared__ __align__(16) bf16 Kt[2][32][72];       // [key][hd]
  __shared__ __align__(16) bf16 VtT[2][64][40];      // [hd][key]
  __shared__ __align__(16) bf16 Ps[4][16][40];
  const int tid = threadIdx.x, lane = tid & 31, w = tid >> 5;
  const int hi = lane >> 4, ln = lane & 15;
  const int kb = hi << 3, kb2 = hi << 4;
  const int bh = blockIdx.x;                         // 0..127
  const int b = bh >> 4, h = bh & 15;
  const int qblk = blockIdx.y;                       // 0..15
  const int q0 = qblk * 64 + w * 16;
  const long base = (long)bh * 65536;

  auto issueKV = [&](int c, int bi) {
    const int j0 = c * 32;
    #pragma unroll
    for (int it = 0; it < 2; ++it) {                 // K tile 32x64
      int i = tid + it * 128;
      int r = i >> 3, cc = (i & 7) * 8;
      cp16(&Kt[bi][r][cc], &Kh[base + (long)(j0 + r) * 64 + cc]);
    }
    #pragma unroll
    for (int it = 0; it < 2; ++it) {                 // V tile (transposed) 64x32
      int i = tid + it * 128;
      int d = i >> 2, cc = (i & 3) * 8;
      cp16(&VtT[bi][d][cc], &VhT[base + (long)d * 1024 + j0 + cc]);
    }
  };

  v16bf qf[2];                                       // Q resident in registers
  {
    const bf16* qrow = Qh + base + (long)(q0 + ln) * 64;
    #pragma unroll
    for (int kc = 0; kc < 2; ++kc)
      qf[kc] = concat8(*(const bf16x8*)(qrow + kc * 32 + kb),
                       *(const bf16x8*)(qrow + kc * 32 + kb + 16));
  }
  v8f o[4] = {};
  float m[8], l[8];
  #pragma unroll
  for (int r = 0; r < 8; ++r) { m[r] = NEG_BIG; l[r] = 0.f; }
  const float scale = 0.03125f;                      // 1/sqrt(1024)
  const int nchunks = qblk * 2 + 2;
  issueKV(0, 0);
  for (int c = 0; c < nchunks; ++c) {
    const int j0 = c * 32, buf = c & 1;
    cp_wait();
    __syncthreads();
    if (j0 <= q0 + 15) {                             // wave-uniform guard
      v8f sc[2];
      #pragma unroll
      for (int t = 0; t < 2; ++t) {
        v8f s = {};
        #pragma unroll
        for (int kc = 0; kc < 2; ++kc) {
          const bf16* kr = &Kt[buf][16 * t + ln][kc * 32 + kb];
          v16bf bfr = concat8(*(const bf16x8*)kr, *(const bf16x8*)(kr + 16));
          s = __builtin_amdgcn_wmma_f32_16x16x32_bf16(false, qf[kc], false, bfr,
                                                      (short)0, s, false, false);
        }
        sc[t] = s;
      }
      #pragma unroll
      for (int r = 0; r < 8; ++r) {
        const int qrow = q0 + (hi << 3) + r;
        #pragma unroll
        for (int t = 0; t < 2; ++t) {
          const int key = j0 + 16 * t + ln;
          float v = sc[t][r] * scale;
          sc[t][r] = (key <= qrow) ? v : NEG_BIG;
        }
        float mx = fmaxf(sc[0][r], sc[1][r]);
        #pragma unroll
        for (int off = 8; off; off >>= 1)
          mx = fmaxf(mx, __shfl_xor(mx, off, 32));
        float mnew = fmaxf(m[r], mx);
        float alpha = __expf(m[r] - mnew);
        m[r] = mnew;
        float ps = 0.f;
        #pragma unroll
        for (int t = 0; t < 2; ++t) {
          float p = __expf(sc[t][r] - mnew);
          sc[t][r] = p;
          ps += p;
        }
        #pragma unroll
        for (int off = 8; off; off >>= 1) ps += __shfl_xor(ps, off, 32);
        l[r] = l[r] * alpha + ps;
        #pragma unroll
        for (int f = 0; f < 4; ++f) o[f][r] *= alpha;
        Ps[w][(hi << 3) + r][ln]      = (bf16)sc[0][r];
        Ps[w][(hi << 3) + r][16 + ln] = (bf16)sc[1][r];
      }
      const bf16* pr = &Ps[w][ln][0];                // LDS in-order within wave
      v16bf pf = concat8(*(const bf16x8*)(pr + kb),
                         *(const bf16x8*)(pr + kb + 16));
      #pragma unroll
      for (int f = 0; f < 4; ++f) {
        const bf16* vr = &VtT[buf][16 * f + ln][kb];
        v16bf vf = concat8(*(const bf16x8*)vr, *(const bf16x8*)(vr + 16));
        o[f] = __builtin_amdgcn_wmma_f32_16x16x32_bf16(false, pf, false, vf,
                                                       (short)0, o[f], false, false);
      }
    }
    __syncthreads();
    if (c + 1 < nchunks) issueKV(c + 1, buf ^ 1);    // DMA next chunk after compute of this one
  }
  #pragma unroll
  for (int r = 0; r < 8; ++r) {
    const float inv = 1.0f / l[r];
    const int s = q0 + (hi << 3) + r;
    #pragma unroll
    for (int f = 0; f < 4; ++f) {
      long idx = ((long)(b * 1024 + s)) * 1024 + h * 64 + 16 * f + ln;
      Obuf[idx] = (bf16)(o[f][r] * inv);
    }
  }
}

// --------------------------------------------- out-proj + residual + LayerNorm
// 16 tokens x full D=1024 per block; 8 waves x 128 cols; WoT staged per 512-col half
__global__ __launch_bounds__(256) void outproj_ln(const bf16* __restrict__ Obuf,
                                                  const bf16* __restrict__ WoT, // [N][K]
                                                  const float* __restrict__ bo,
                                                  const float* __restrict__ x,
                                                  const float* __restrict__ gamma,
                                                  const float* __restrict__ beta,
                                                  float* __restrict__ out) {
  __shared__ __align__(16) bf16 As[16][40];
  __shared__ __align__(16) bf16 BsT[512][40];        // N-major: 512 cols x 32 K
  __shared__ float redS[8][16], redQ[8][16];
  const int tid = threadIdx.x, lane = tid & 31, w = tid >> 5;
  const int hi = lane >> 4, ln = lane & 15;
  const int kb = hi << 3, kb2 = hi << 4;
  const int m0 = blockIdx.x * 16;
  v8f acc[8] = {};
  for (int k0 = 0; k0 < 1024; k0 += 32) {
    if (tid < 64) {                                  // A: 16x32, vec8
      int r = tid >> 2, c = (tid & 3) * 8;
      *(bf16x8*)&As[r][c] = *(const bf16x8*)&Obuf[(long)(m0 + r) * 1024 + k0 + c];
    }
    #pragma unroll
    for (int half = 0; half < 2; ++half) {
      #pragma unroll
      for (int it = 0; it < 8; ++it) {               // B half: 512 rows x 32 K, async
        int i = tid + it * 256;
        int n = i >> 2, c = (i & 3) * 8;
        cp16(&BsT[n][c], &WoT[(long)(half * 512 + n) * 1024 + k0 + c]);
      }
      cp_wait();
      __syncthreads();
      const bf16* ar = &As[ln][kb];
      v16bf a = concat8(*(const bf16x8*)ar, *(const bf16x8*)(ar + 16));
      #pragma unroll
      for (int f = 0; f < 4; ++f) {
        const bf16* br = &BsT[w * 64 + f * 16 + ln][kb];
        v16bf bfr = concat8(*(const bf16x8*)br, *(const bf16x8*)(br + 16));
        acc[f + 4 * half] = __builtin_amdgcn_wmma_f32_16x16x32_bf16(
            false, a, false, bfr, (short)0, acc[f + 4 * half], false, false);
      }
      __syncthreads();
    }
  }
  float vals[8][8], psum[8], psq[8];
  #pragma unroll
  for (int r = 0; r < 8; ++r) { psum[r] = 0.f; psq[r] = 0.f; }
  #pragma unroll
  for (int f = 0; f < 8; ++f) {
    const int col = (f >> 2) * 512 + w * 64 + (f & 3) * 16 + ln;
    const float bias = bo[col];
    #pragma unroll
    for (int r = 0; r < 8; ++r) {
      const int row = m0 + (hi << 3) + r;
      float v = acc[f][r] + bias + x[(long)row * 1024 + col];
      vals[f][r] = v;
      psum[r] += v;
      psq[r] += v * v;
    }
  }
  #pragma unroll
  for (int r = 0; r < 8; ++r)
    #pragma unroll
    for (int off = 8; off; off >>= 1) {
      psum[r] += __shfl_xor(psum[r], off, 32);
      psq[r]  += __shfl_xor(psq[r], off, 32);
    }
  if (ln == 0) {
    #pragma unroll
    for (int r = 0; r < 8; ++r) {
      redS[w][(hi << 3) + r] = psum[r];
      redQ[w][(hi << 3) + r] = psq[r];
    }
  }
  __syncthreads();
  float mu[8], rs[8];
  #pragma unroll
  for (int r = 0; r < 8; ++r) {
    const int row = (hi << 3) + r;
    float S = 0.f, Q = 0.f;
    #pragma unroll
    for (int ww = 0; ww < 8; ++ww) { S += redS[ww][row]; Q += redQ[ww][row]; }
    mu[r] = S * (1.0f / 1024.0f);
    float var = Q * (1.0f / 1024.0f) - mu[r] * mu[r];
    rs[r] = rsqrtf(var + 1e-5f);
  }
  #pragma unroll
  for (int f = 0; f < 8; ++f) {
    const int col = (f >> 2) * 512 + w * 64 + (f & 3) * 16 + ln;
    const float g = gamma[col], bb = beta[col];
    #pragma unroll
    for (int r = 0; r < 8; ++r) {
      const int row = m0 + (hi << 3) + r;
      out[(long)row * 1024 + col] = (vals[f][r] - mu[r]) * rs[r] * g + bb;
    }
  }
}

// ---------------------------------------------------------------- launch
extern "C" void kernel_launch(void* const* d_in, const int* in_sizes, int n_in,
                              void* d_out, int out_size, void* d_ws, size_t ws_size,
                              hipStream_t stream) {
  const float* x     = (const float*)d_in[0];
  const float* Wqkv  = (const float*)d_in[1];
  const float* bqkv  = (const float*)d_in[2];
  const float* Wout  = (const float*)d_in[3];
  const float* bout  = (const float*)d_in[4];
  const float* gamma = (const float*)d_in[5];
  const float* beta  = (const float*)d_in[6];
  float* out = (float*)d_out;

  char* ws = (char*)d_ws;
  bf16* WqT  = (bf16*)(ws + 0);          //  6,291,456 B  [3072][1024] col-major
  bf16* WoT  = (bf16*)(ws + 6291456);    //  2,097,152 B  [1024][1024] col-major
  bf16* Qh   = (bf16*)(ws + 8388608);    // 16,777,216 B  [bh][S][64]
  bf16* Kh   = (bf16*)(ws + 25165824);   // 16,777,216 B  [bh][S][64]
  bf16* VhT  = (bf16*)(ws + 41943040);   // 16,777,216 B  [bh][64][S] transposed
  bf16* Obuf = (bf16*)(ws + 58720256);   // 16,777,216 B  [token][D]

  cvt_transpose<<<(3145728 + 255) / 256, 256, 0, stream>>>(Wqkv, WqT, 1024, 3072);
  cvt_transpose<<<(1048576 + 255) / 256, 256, 0, stream>>>(Wout, WoT, 1024, 1024);
  qkv_gemm<<<dim3(8192 / TM, 3072 / TN), 256, 0, stream>>>(x, WqT, bqkv, Qh, Kh, VhT);
  attn_fwd<<<dim3(128, 16), 128, 0, stream>>>(Qh, Kh, VhT, Obuf);
  outproj_ln<<<512, 256, 0, stream>>>(Obuf, WoT, bout, x, gamma, beta, out);
}
